// MultiHeadAttention_57363583205864
// MI455X (gfx1250) — matrix operand order repaired and checked by reading, not compiled, not run on hardware
//
#include <hip/hip_runtime.h>

#define DMODEL 1024
#define SEQ    1024
#define NBAT   2
#define NHEAD  16
#define DHEAD  64

typedef __attribute__((ext_vector_type(16))) __bf16 bf16x16;
typedef __attribute__((ext_vector_type(8)))  float  f32x8;

struct U16x16 { uint4 lo; uint4 hi; };

// fp32 -> bf16, round-to-nearest-even
static __device__ __forceinline__ unsigned short f2bf(float f) {
  unsigned int u = __builtin_bit_cast(unsigned int, f);
  u += 0x7fffu + ((u >> 16) & 1u);
  return (unsigned short)(u >> 16);
}

// A-fragment (16x32 bf16): lanes 0-15 row M=lane, elems 0-7 = K(hv*8..+7),
// elems 8-15 = K(16+hv*8..+7); hv = lane>>4 selects the interleaved K half.
static __device__ __forceinline__ bf16x16 ldfrag_a(const unsigned short* p, int hv) {
  U16x16 t;
  t.lo = *(const uint4*)(p + hv * 8);
  t.hi = *(const uint4*)(p + 16 + hv * 8);
  return __builtin_bit_cast(bf16x16, t);
}
// B-fragment (32x16 bf16): lane = column N (lane&15), elems 0-7 = K(hv*8..+7),
// elems 8-15 = K(16+hv*8..+7), matching the A-fragment K order.
static __device__ __forceinline__ bf16x16 ldfrag_b(const unsigned short* p, int hv) {
  U16x16 t;
  t.lo = *(const uint4*)(p + hv * 8);
  t.hi = *(const uint4*)(p + 16 + hv * 8);
  return __builtin_bit_cast(bf16x16, t);
}

static __device__ __forceinline__ f32x8 wmma_bf(bf16x16 a, bf16x16 b, f32x8 c) {
  return __builtin_amdgcn_wmma_f32_16x16x32_bf16(false, a, false, b, (short)0, c,
                                                 false, false);
}

// reductions across the 16 lanes of a half-wave (C-layout row lives there)
static __device__ __forceinline__ float hmax16(float v) {
  v = fmaxf(v, __shfl_xor(v, 1));
  v = fmaxf(v, __shfl_xor(v, 2));
  v = fmaxf(v, __shfl_xor(v, 4));
  v = fmaxf(v, __shfl_xor(v, 8));
  return v;
}
static __device__ __forceinline__ float hsum16(float v) {
  v += __shfl_xor(v, 1);
  v += __shfl_xor(v, 2);
  v += __shfl_xor(v, 4);
  v += __shfl_xor(v, 8);
  return v;
}

// fp32 -> bf16 conversion, 4 elements per thread
__global__ void cvt_bf16(const float* __restrict__ s, unsigned short* __restrict__ d,
                         int n4) {
  int i = blockIdx.x * blockDim.x + threadIdx.x;
  if (i < n4) {
    float4 f = ((const float4*)s)[i];
    ushort4 o;
    o.x = f2bf(f.x);
    o.y = f2bf(f.y);
    o.z = f2bf(f.z);
    o.w = f2bf(f.w);
    ((ushort4*)d)[i] = o;
  }
}

// C = A(M x 1024) @ W(N x 1024)^T + bias. Wave tile 32M x 64N.
// K-loop is 2x unrolled with ping-pong fragment sets X/Y: load Y -> consume X ->
// load X -> consume Y, so no register rotation copies and loads overlap WMMAs.
// mode 0: bf16 (b,h,s,dh) layout, mode 1: bf16 (b,h,dh,s) (for V),
// mode 2: fp32 flat (final output projection).
__global__ __launch_bounds__(128) void gemm_bf16(
    const unsigned short* __restrict__ A, const unsigned short* __restrict__ W,
    const float* __restrict__ bias, void* __restrict__ out, int mode) {
  const int lane = threadIdx.x & 31;
  const int wave = threadIdx.x >> 5;
  const int hv = lane >> 4;
  const int l15 = lane & 15;
  const int m0 = blockIdx.y * 128 + wave * 32;
  const int n0 = blockIdx.x * 64;

  f32x8 acc[2][4];
#pragma unroll
  for (int mt = 0; mt < 2; ++mt)
#pragma unroll
    for (int n = 0; n < 4; ++n)
#pragma unroll
      for (int r = 0; r < 8; ++r) acc[mt][n][r] = 0.f;

  const unsigned short* arow0 = A + (size_t)(m0 + l15) * DMODEL;
  const unsigned short* arow1 = arow0 + (size_t)16 * DMODEL;
  const unsigned short* wrow[4];
#pragma unroll
  for (int n = 0; n < 4; ++n) wrow[n] = W + (size_t)(n0 + n * 16 + l15) * DMODEL;

  // buffer X <- chunk 0
  bf16x16 a0x = ldfrag_a(arow0, hv);
  bf16x16 a1x = ldfrag_a(arow1, hv);
  bf16x16 bx[4];
#pragma unroll
  for (int n = 0; n < 4; ++n) bx[n] = ldfrag_b(wrow[n], hv);

  for (int k0 = 0; k0 < DMODEL; k0 += 64) {
    // buffer Y <- chunk k0+32 (always in range: k0 <= DMODEL-64)
    bf16x16 a0y = ldfrag_a(arow0 + k0 + 32, hv);
    bf16x16 a1y = ldfrag_a(arow1 + k0 + 32, hv);
    bf16x16 by[4];
#pragma unroll
    for (int n = 0; n < 4; ++n) by[n] = ldfrag_b(wrow[n] + k0 + 32, hv);

    // consume X (chunk k0)
#pragma unroll
    for (int n = 0; n < 4; ++n) acc[0][n] = wmma_bf(a0x, bx[n], acc[0][n]);
#pragma unroll
    for (int n = 0; n < 4; ++n) acc[1][n] = wmma_bf(a1x, bx[n], acc[1][n]);

    // buffer X <- chunk k0+64 (skip past end)
    if (k0 + 64 < DMODEL) {
      a0x = ldfrag_a(arow0 + k0 + 64, hv);
      a1x = ldfrag_a(arow1 + k0 + 64, hv);
#pragma unroll
      for (int n = 0; n < 4; ++n) bx[n] = ldfrag_b(wrow[n] + k0 + 64, hv);
    }

    // consume Y (chunk k0+32)
#pragma unroll
    for (int n = 0; n < 4; ++n) acc[0][n] = wmma_bf(a0y, by[n], acc[0][n]);
#pragma unroll
    for (int n = 0; n < 4; ++n) acc[1][n] = wmma_bf(a1y, by[n], acc[1][n]);
  }

#pragma unroll
  for (int mt = 0; mt < 2; ++mt) {
#pragma unroll
    for (int n = 0; n < 4; ++n) {
      const int nc = n0 + n * 16 + l15;
      const float bv = bias[nc];
#pragma unroll
      for (int r = 0; r < 8; ++r) {
        const int m = m0 + mt * 16 + r + 8 * hv;
        const float o = acc[mt][n][r] + bv;
        if (mode == 2) {
          ((float*)out)[(size_t)m * DMODEL + nc] = o;
        } else {
          const int bb = m >> 10, s = m & (SEQ - 1);
          const int h = nc >> 6, dh = nc & (DHEAD - 1);
          const unsigned short v = f2bf(o);
          if (mode == 0)
            ((unsigned short*)out)[((size_t)(bb * NHEAD + h) * SEQ + s) * DHEAD + dh] = v;
          else
            ((unsigned short*)out)[((size_t)(bb * NHEAD + h) * DHEAD + dh) * SEQ + s] = v;
        }
      }
    }
  }
}

// Fused DeBERTa attention, flash style. Workgroup = (b,h) x 64 query rows.
// scores[i,j] = (q_i.k_j + q_i.kr_pos + qr_pos.k_j)/sqrt(192), pos=clamp(i-j+512).
__global__ __launch_bounds__(128) void deberta_attn(
    const unsigned short* __restrict__ qb, const unsigned short* __restrict__ kb,
    const unsigned short* __restrict__ qrb, const unsigned short* __restrict__ krb,
    const unsigned short* __restrict__ vtb, const int* __restrict__ pmask,
    unsigned short* __restrict__ attn_out) {
  __shared__ float sP2C[128 * 64];            // workgroup-shared qr-band x K   (32 KB)
  __shared__ float sC2P[4][16 * 80];          // per-wave q x kr-band           (20 KB)
  __shared__ unsigned short sP[4][16 * 72];   // per-wave P transpose, padded   (9 KB)

  const int bh = blockIdx.y;
  const int bb = bh / NHEAD;
  const int hh = bh % NHEAD;
  const int lane = threadIdx.x & 31;
  const int wave = threadIdx.x >> 5;
  const int hv = lane >> 4;
  const int l15 = lane & 15;

  const int I0 = blockIdx.x * 64;   // workgroup query base
  const int i0 = I0 + wave * 16;    // wave query base

  const unsigned short* qh = qb + (size_t)bh * SEQ * DHEAD;
  const unsigned short* kh = kb + (size_t)bh * SEQ * DHEAD;
  const unsigned short* qrh = qrb + (size_t)bh * SEQ * DHEAD;
  const unsigned short* krh = krb + (size_t)bh * SEQ * DHEAD;
  const unsigned short* vth = vtb + (size_t)bh * DHEAD * SEQ;
  const int* pm = pmask + bb * SEQ;

  // Q fragments are loop-invariant (K=64 -> two K=32 chunks)
  bf16x16 qa0, qa1;
  {
    const unsigned short* qrow = qh + (size_t)(i0 + l15) * DHEAD;
    qa0 = ldfrag_a(qrow, hv);
    qa1 = ldfrag_a(qrow + 32, hv);
  }

  f32x8 acc[4];
  float mrow[8], lrow[8];
#pragma unroll
  for (int r = 0; r < 8; ++r) {
    mrow[r] = -__builtin_inff();
    lrow[r] = 0.f;
  }
#pragma unroll
  for (int n = 0; n < 4; ++n)
#pragma unroll
    for (int r = 0; r < 8; ++r) acc[n][r] = 0.f;

  const float inv_scale = 0.0721687836f;  // 1/sqrt(3*DH)

  for (int j0 = 0; j0 < SEQ; j0 += 64) {
    // ---- prefetch next key step's K rows and V^T rows into cache ----
    if (j0 + 64 < SEQ) {
      const int pr = threadIdx.x & 63;  // 128 threads cover 64 rows twice
      __builtin_prefetch(kh + (size_t)(j0 + 64 + pr) * DHEAD, 0, 1);
      __builtin_prefetch(vth + (size_t)pr * SEQ + j0 + 64, 0, 1);
    }

    // ---- K-tile B-fragments, shared by c2c and the p2c band ----
    bf16x16 kf0[4], kf1[4];
#pragma unroll
    for (int n = 0; n < 4; ++n) {
      const unsigned short* bcol = kh + (size_t)(j0 + n * 16 + l15) * DHEAD;
      kf0[n] = ldfrag_b(bcol, hv);
      kf1[n] = ldfrag_b(bcol + 32, hv);
    }

    // ---- c2c scores ----
    f32x8 sc[4];
#pragma unroll
    for (int n = 0; n < 4; ++n) {
      f32x8 c;
#pragma unroll
      for (int r = 0; r < 8; ++r) c[r] = 0.f;
      c = wmma_bf(qa0, kf0[n], c);
      c = wmma_bf(qa1, kf1[n], c);
      sc[n] = c;
    }

    // ---- cooperative p2c band: p2c_att[p][j] = qr_p . k_j over 128-row band ----
    const int Dw = I0 - j0 + 512;
    int P0 = Dw - 63;
    P0 = P0 < 0 ? 0 : (P0 > SEQ - 128 ? SEQ - 128 : P0);
#pragma unroll
    for (int tt = 0; tt < 2; ++tt) {
      const int prow = P0 + wave * 32 + tt * 16 + l15;
      const unsigned short* arow = qrh + (size_t)prow * DHEAD;
      bf16x16 a0 = ldfrag_a(arow, hv);
      bf16x16 a1 = ldfrag_a(arow + 32, hv);
#pragma unroll
      for (int n = 0; n < 4; ++n) {
        f32x8 c;
#pragma unroll
        for (int r = 0; r < 8; ++r) c[r] = 0.f;
        c = wmma_bf(a0, kf0[n], c);
        c = wmma_bf(a1, kf1[n], c);
#pragma unroll
        for (int r = 0; r < 8; ++r)
          sP2C[(wave * 32 + tt * 16 + r + 8 * hv) * 64 + n * 16 + l15] = c[r];
      }
    }
    __syncthreads();

    // ---- per-wave c2p band: c2p_att[i][c] = q_i . kr[p0w+c], 80-wide ----
    const int d = i0 - j0 + 512;
    int p0w = d - 63;
    p0w = p0w < 0 ? 0 : (p0w > SEQ - 80 ? SEQ - 80 : p0w);
#pragma unroll
    for (int t = 0; t < 5; ++t) {
      const unsigned short* bcol = krh + (size_t)(p0w + t * 16 + l15) * DHEAD;
      f32x8 c;
#pragma unroll
      for (int r = 0; r < 8; ++r) c[r] = 0.f;
      c = wmma_bf(qa0, ldfrag_b(bcol, hv), c);
      c = wmma_bf(qa1, ldfrag_b(bcol + 32, hv), c);
#pragma unroll
      for (int r = 0; r < 8; ++r)
        sC2P[wave][(r + 8 * hv) * 80 + t * 16 + l15] = c[r];
    }

    // ---- band gathers + scale + padding mask ----
#pragma unroll
    for (int n = 0; n < 4; ++n) {
      const int j = j0 + n * 16 + l15;
      const float madd = (pm[j] == 1) ? -__builtin_inff() : 0.f;
#pragma unroll
      for (int r = 0; r < 8; ++r) {
        const int i = i0 + r + 8 * hv;
        int pos = i - j + 512;
        pos = pos < 0 ? 0 : (pos > SEQ - 1 ? SEQ - 1 : pos);
        const float v1 = sC2P[wave][(r + 8 * hv) * 80 + (pos - p0w)];
        const float v2 = sP2C[(pos - P0) * 64 + n * 16 + l15];
        sc[n][r] = (sc[n][r] + v1 + v2) * inv_scale + madd;
      }
    }

    // ---- online softmax ----
#pragma unroll
    for (int r = 0; r < 8; ++r) {
      float v = fmaxf(fmaxf(sc[0][r], sc[1][r]), fmaxf(sc[2][r], sc[3][r]));
      const float mnew = fmaxf(mrow[r], hmax16(v));
      const float sf = __expf(mrow[r] - mnew);
      float ls = 0.f;
#pragma unroll
      for (int n = 0; n < 4; ++n) {
        const float p = __expf(sc[n][r] - mnew);
        sc[n][r] = p;
        ls += p;
      }
      ls = hsum16(ls);
      lrow[r] = lrow[r] * sf + ls;
      mrow[r] = mnew;
#pragma unroll
      for (int n = 0; n < 4; ++n) acc[n][r] *= sf;
    }

    // ---- P -> LDS (C-layout to A-layout transpose), padded stride 72 ----
#pragma unroll
    for (int n = 0; n < 4; ++n)
#pragma unroll
      for (int r = 0; r < 8; ++r)
        sP[wave][(r + 8 * hv) * 72 + n * 16 + l15] = f2bf(sc[n][r]);

    // ---- acc += P @ V (V^T layout gives contiguous B-fragment reads) ----
    {
      const unsigned short* pb = &sP[wave][0] + (size_t)l15 * 72;
      bf16x16 pa0 = ldfrag_a(pb, hv);
      bf16x16 pa1 = ldfrag_a(pb + 32, hv);
#pragma unroll
      for (int n = 0; n < 4; ++n) {
        const unsigned short* vcol = vth + (size_t)(n * 16 + l15) * SEQ + j0;
        acc[n] = wmma_bf(pa0, ldfrag_b(vcol, hv), acc[n]);
        acc[n] = wmma_bf(pa1, ldfrag_b(vcol + 32, hv), acc[n]);
      }
    }
    __syncthreads();  // protect sP2C for next key step
  }

  // ---- normalize and store (b, s, D) bf16 for the output projection ----
#pragma unroll
  for (int n = 0; n < 4; ++n) {
    const int nc = hh * DHEAD + n * 16 + l15;
#pragma unroll
    for (int r = 0; r < 8; ++r) {
      const int i = i0 + r + 8 * hv;
      const float o = acc[n][r] / lrow[r];
      attn_out[(size_t)(bb * SEQ + i) * DMODEL + nc] = f2bf(o);
    }
  }
}

extern "C" void kernel_launch(void* const* d_in, const int* in_sizes, int n_in,
                              void* d_out, int out_size, void* d_ws, size_t ws_size,
                              hipStream_t stream) {
  (void)in_sizes; (void)n_in; (void)out_size; (void)ws_size;
  const float* x = (const float*)d_in[0];
  const float* rel = (const float*)d_in[1];
  const int* pmask = (const int*)d_in[2];
  const float* Wq = (const float*)d_in[3];
  const float* bq = (const float*)d_in[4];
  const float* Wk = (const float*)d_in[5];
  const float* bk = (const float*)d_in[6];
  const float* Wv = (const float*)d_in[7];
  const float* bv = (const float*)d_in[8];
  const float* Wqr = (const float*)d_in[9];
  const float* bqr = (const float*)d_in[10];
  const float* Wkr = (const float*)d_in[11];
  const float* bkr = (const float*)d_in[12];
  const float* Wc = (const float*)d_in[13];
  const float* bc = (const float*)d_in[14];

  const size_t NX = (size_t)NBAT * SEQ * DMODEL;  // 2M elems
  const size_t NW = (size_t)DMODEL * DMODEL;      // 1M elems

  char* base = (char*)d_ws;
  size_t off = 0;
  auto take = [&](size_t elems) {
    unsigned short* p = (unsigned short*)(base + off);
    off += elems * sizeof(unsigned short);
    off = (off + 255) & ~(size_t)255;
    return p;
  };
  unsigned short* xb = take(NX);
  unsigned short* rb = take(NX);
  unsigned short* wqb = take(NW);
  unsigned short* wkb = take(NW);
  unsigned short* wvb = take(NW);
  unsigned short* wqrb = take(NW);
  unsigned short* wkrb = take(NW);
  unsigned short* wcb = take(NW);
  unsigned short* qbuf = take(NX);
  unsigned short* kbuf = take(NX);
  unsigned short* qrbf = take(NX);
  unsigned short* krbf = take(NX);
  unsigned short* vtbf = take(NX);
  unsigned short* atb = take(NX);

  const int ct = 256;
  const int gx = (int)((NX / 4 + ct - 1) / ct);
  const int gw = (int)((NW / 4 + ct - 1) / ct);
  cvt_bf16<<<gx, ct, 0, stream>>>(x, xb, (int)(NX / 4));
  cvt_bf16<<<gx, ct, 0, stream>>>(rel, rb, (int)(NX / 4));
  cvt_bf16<<<gw, ct, 0, stream>>>(Wq, wqb, (int)(NW / 4));
  cvt_bf16<<<gw, ct, 0, stream>>>(Wk, wkb, (int)(NW / 4));
  cvt_bf16<<<gw, ct, 0, stream>>>(Wv, wvb, (int)(NW / 4));
  cvt_bf16<<<gw, ct, 0, stream>>>(Wqr, wqrb, (int)(NW / 4));
  cvt_bf16<<<gw, ct, 0, stream>>>(Wkr, wkrb, (int)(NW / 4));
  cvt_bf16<<<gw, ct, 0, stream>>>(Wc, wcb, (int)(NW / 4));

  dim3 gg(DMODEL / 64, (NBAT * SEQ) / 128);  // (16, 16)
  gemm_bf16<<<gg, 128, 0, stream>>>(xb, wqb, bq, qbuf, 0);
  gemm_bf16<<<gg, 128, 0, stream>>>(xb, wkb, bk, kbuf, 0);
  gemm_bf16<<<gg, 128, 0, stream>>>(xb, wvb, bv, vtbf, 1);
  gemm_bf16<<<gg, 128, 0, stream>>>(rb, wqrb, bqr, qrbf, 0);
  gemm_bf16<<<gg, 128, 0, stream>>>(rb, wkrb, bkr, krbf, 0);

  deberta_attn<<<dim3(SEQ / 64, NBAT * NHEAD), 128, 0, stream>>>(
      qbuf, kbuf, qrbf, krbf, vtbf, pmask, atb);

  gemm_bf16<<<gg, 128, 0, stream>>>(atb, wcb, bc, d_out, 2);
}
